// Attention_65541200937161
// MI455X (gfx1250) — matrix operand order repaired and checked by reading, not compiled, not run on hardware
//
#include <hip/hip_runtime.h>
#include <hip/hip_bf16.h>

typedef __attribute__((ext_vector_type(16))) _Float16 v16h;
typedef __attribute__((ext_vector_type(8)))  _Float16 v8h;
typedef __attribute__((ext_vector_type(8)))  float    v8f;
typedef __attribute__((ext_vector_type(4)))  int      v4i;

#define BB  16
#define NNQ 1024
#define EE  512
#define HHh 8
#define DHH 64

// LDS row pitches (halves) chosen so 16 consecutive rows hit distinct banks
// while keeping 16B alignment for b128 ds accesses.
#define KROW 72
#define VROW 40
#define PROW 40
#define KBUFSZ (32 * KROW)
#define VBUFSZ (64 * VROW)

#if defined(__has_builtin)
#if __has_builtin(__builtin_amdgcn_global_load_async_to_lds_b128) && \
    __has_builtin(__builtin_amdgcn_s_wait_asynccnt)
#define USE_ASYNC 1
#endif
#endif
#ifndef USE_ASYNC
#define USE_ASYNC 0
#endif

#define AS1 __attribute__((address_space(1)))
#define AS3 __attribute__((address_space(3)))

__device__ __forceinline__ v16h combine8(v8h lo, v8h hi) {
  return __builtin_shufflevector(lo, hi, 0,1,2,3,4,5,6,7,8,9,10,11,12,13,14,15);
}

// A-fragment (16x32 f16): lane holds row (lane&15), 16 consecutive K starting
// at 2*base0, base0=(lane>>4)*8 -> two 16B loads.
__device__ __forceinline__ v16h load_frag_h(const _Float16* p, int base0) {
  v8h lo = *(const v8h*)(p + 2 * base0);
  v8h hi = *(const v8h*)(p + 2 * base0 + 8);
  return combine8(lo, hi);
}

// B-fragment: lane holds col (lane&15), 16 consecutive K starting at kb.
__device__ __forceinline__ v16h frag16(const _Float16* p) {
  v8h lo = *(const v8h*)p;
  v8h hi = *(const v8h*)(p + 8);
  return combine8(lo, hi);
}

__device__ __forceinline__ v8f wmma_f16(v16h a, v16h b, v8f c) {
  return __builtin_amdgcn_wmma_f32_16x16x32_f16(false, a, false, b, (short)0, c,
                                                false, false);
}

__device__ __forceinline__ float rmax16(float x) {
  x = fmaxf(x, __shfl_xor(x, 1, 32));
  x = fmaxf(x, __shfl_xor(x, 2, 32));
  x = fmaxf(x, __shfl_xor(x, 4, 32));
  x = fmaxf(x, __shfl_xor(x, 8, 32));
  return x;
}
__device__ __forceinline__ float rsum16(float x) {
  x += __shfl_xor(x, 1, 32);
  x += __shfl_xor(x, 2, 32);
  x += __shfl_xor(x, 4, 32);
  x += __shfl_xor(x, 8, 32);
  return x;
}

// Stage one 16B chunk of K-tile and one of V-tile for this thread.
__device__ __forceinline__ void stage_chunks(const _Float16* gk, const _Float16* gv,
                                             _Float16* lk, _Float16* lv) {
#if USE_ASYNC
  __builtin_amdgcn_global_load_async_to_lds_b128((AS1 v4i*)gk, (AS3 v4i*)lk, 0, 0);
  __builtin_amdgcn_global_load_async_to_lds_b128((AS1 v4i*)gv, (AS3 v4i*)lv, 0, 0);
#else
  *(v8h*)lk = *(const v8h*)gk;
  *(v8h*)lv = *(const v8h*)gv;
#endif
}

template <int N>
__device__ __forceinline__ void wait_async() {
#if USE_ASYNC
  __builtin_amdgcn_s_wait_asynccnt(N);
#endif
}

// ---------------- weight fp32 -> f16 convert (4 matrices, contiguous dest) ---
__global__ void wcvt_kernel(const float* __restrict__ Wq, const float* __restrict__ Wk,
                            const float* __restrict__ Wv, const float* __restrict__ Wp,
                            _Float16* __restrict__ dst) {
  int t = blockIdx.x * blockDim.x + threadIdx.x;   // 0 .. 4*262144
  int sel = t >> 18;
  int i = t & 262143;
  const float* s = (sel == 0) ? Wq : (sel == 1) ? Wk : (sel == 2) ? Wv : Wp;
  dst[t] = (_Float16)s[i];
}

// ---------------- input projection: Y = X * W^T (+bias), f16 output ---------
// vmode 0: plain layout y[(b*N+n)*E + f]      (used for Q, K)
// vmode 1: per-head transposed Vt[((b*H+h)*DH + d)*N + n]   (used for V)
__global__ void proj_kernel(const float* __restrict__ X, const _Float16* __restrict__ W,
                            const float* __restrict__ bias, _Float16* __restrict__ dst,
                            int vmode) {
  const int lane  = threadIdx.x & 31;
  const int wave  = threadIdx.x >> 5;
  const int mtile = blockIdx.x * 8 + wave;      // [0,1024)
  const int row0  = mtile * 16;                 // row in flattened (B*N, E)
  const int fs    = blockIdx.y * 64;            // output-feature strip
  const int base0 = (lane >> 4) * 8;
  const int kb    = (lane >> 4) * 16;
  const int colA  = lane & 15;

  v8f z = {};
  v8f c[4];
  c[0] = z; c[1] = z; c[2] = z; c[3] = z;

  const float* arow = X + (size_t)(row0 + colA) * EE;

  for (int k = 0; k < EE; k += 32) {
    v16h a;
    const float* ap = arow + k + kb;
#pragma unroll
    for (int j = 0; j < 8; ++j) a[j] = (_Float16)ap[j];
#pragma unroll
    for (int j = 0; j < 8; ++j) a[8 + j] = (_Float16)ap[8 + j];
#pragma unroll
    for (int ft = 0; ft < 4; ++ft) {
      int f = fs + ft * 16 + colA;
      v16h b = *(const v16h*)(W + (size_t)f * EE + k + kb);
      c[ft] = wmma_f16(a, b, c[ft]);
    }
  }

#pragma unroll
  for (int ft = 0; ft < 4; ++ft) {
    int f = fs + ft * 16 + colA;
    float bv = bias[f];
    if (vmode == 0) {
#pragma unroll
      for (int v = 0; v < 8; ++v) {
        int r = row0 + (lane >> 4) * 8 + v;
        dst[(size_t)r * EE + f] = (_Float16)(c[ft][v] + bv);
      }
    } else {
      int h = f >> 6, dd = f & 63;
#pragma unroll
      for (int v = 0; v < 8; ++v) {
        int r = row0 + (lane >> 4) * 8 + v;
        int bidx = r >> 10, n = r & 1023;
        dst[(size_t)((bidx * HHh + h) * DHH + dd) * NNQ + n] =
            (_Float16)(c[ft][v] + bv);
      }
    }
  }
}

// ---------------- fused attention: flash-style one pass ---------------------
// out = ((exp(S-m)/Z) * d) @ V  with S = QK^T*scale + d, Z = rowsum(exp(S-m))
// All 8 waves of a block share (b,h); K/V tiles are cooperatively staged into
// LDS with double-buffered async copies (GLOBAL_LOAD_ASYNC_TO_LDS_B128).
__global__ void attn_kernel(const _Float16* __restrict__ Q16,
                            const _Float16* __restrict__ K16,
                            const _Float16* __restrict__ Vt,
                            const float* __restrict__ dmat,
                            _Float16* __restrict__ Xb) {
  __shared__ __align__(16) _Float16 Kb[2 * KBUFSZ];   // 2 x (32 x 72)  = 9.0 KB
  __shared__ __align__(16) _Float16 Vb[2 * VBUFSZ];   // 2 x (64 x 40)  = 10.0 KB
  __shared__ __align__(16) _Float16 Pb[8 * 16 * PROW];// per-wave P tile, 10.0 KB

  const int tid   = threadIdx.x;
  const int lane  = tid & 31;
  const int wave  = tid >> 5;
  const int b     = blockIdx.x >> 3;
  const int h     = blockIdx.x & 7;
  const int n0    = (blockIdx.y * 8 + wave) * 16;
  const int base0 = (lane >> 4) * 8;
  const int kb    = (lane >> 4) * 16;
  const int col   = lane & 15;
  _Float16* pl = Pb + wave * 16 * PROW;

  const float* dbase = dmat + (size_t)b * NNQ * NNQ;
  const _Float16* krowbase = K16 + (size_t)(b * NNQ) * EE + h * DHH;
  const _Float16* vbase = Vt + (size_t)(b * HHh + h) * DHH * NNQ;

  // per-thread staging chunk: K tile is 32 rows x 64 halves (8 segs of 16B),
  // V tile is 64 rows x 32 halves (4 segs of 16B); 256 threads cover each.
  const int krow_t = tid >> 3, kseg = (tid & 7) * 8;
  const int vrow_t = tid >> 2, vseg = (tid & 3) * 8;
  const _Float16* gK0 = krowbase + (size_t)krow_t * EE + kseg;  // + m0*EE
  const _Float16* gV0 = vbase + (size_t)vrow_t * NNQ + vseg;    // + m0
  _Float16* lK = Kb + krow_t * KROW + kseg;                     // + buf*KBUFSZ
  _Float16* lV = Vb + vrow_t * VROW + vseg;                     // + buf*VBUFSZ

  // Q A-fragments for k = 0..31 and 32..63 (DH = 64), kept in registers
  const _Float16* qrow = Q16 + (size_t)(b * NNQ + n0 + col) * EE + h * DHH;
  v16h aq0 = load_frag_h(qrow, base0);
  v16h aq1 = load_frag_h(qrow + 32, base0);

  v8f z = {};
  v8f o[4];
  o[0] = z; o[1] = z; o[2] = z; o[3] = z;
  float m_old[8], l_sum[8];
#pragma unroll
  for (int v = 0; v < 8; ++v) { m_old[v] = -1e30f; l_sum[v] = 0.0f; }

  const float scale = 0.125f;

  // prologue: stage tile 0 into buffer 0
  stage_chunks(gK0, gV0, lK, lV);

  for (int it = 0; it < 32; ++it) {
    const int m0 = it * 32;
    const int cur = it & 1;

    // stream next tile into the other buffer while we compute on this one
    if (it + 1 < 32) {
      stage_chunks(gK0 + (size_t)(m0 + 32) * EE, gV0 + (m0 + 32),
                   lK + (cur ^ 1) * KBUFSZ, lV + (cur ^ 1) * VBUFSZ);
      wait_async<2>();   // retire the older (current) tile's 2 copies
    } else {
      wait_async<0>();
    }
    __syncthreads();     // current tile visible to all waves

    // --- scores: two 16x16 tiles over keys [m0, m0+32) from LDS ---
    const _Float16* kp0 = Kb + cur * KBUFSZ + col * KROW;
    const _Float16* kp1 = kp0 + 16 * KROW;
    v8f s0 = z, s1 = z;
    {
      v16h b00 = frag16(kp0 + kb);
      v16h b01 = frag16(kp0 + 32 + kb);
      v16h b10 = frag16(kp1 + kb);
      v16h b11 = frag16(kp1 + 32 + kb);
      s0 = wmma_f16(aq0, b00, s0);
      s0 = wmma_f16(aq1, b01, s0);
      s1 = wmma_f16(aq0, b10, s1);
      s1 = wmma_f16(aq1, b11, s1);
    }

    // --- bias add + online softmax (d kept in regs, reused for P*d) ---
    float e0[8], e1[8], db0[8], db1[8], alpha[8];
    int c0 = m0 + col, c1 = c0 + 16;
#pragma unroll
    for (int v = 0; v < 8; ++v) {
      int r = n0 + (lane >> 4) * 8 + v;
      db0[v] = dbase[(size_t)r * NNQ + c0];
      db1[v] = dbase[(size_t)r * NNQ + c1];
      e0[v] = s0[v] * scale + db0[v];
      e1[v] = s1[v] * scale + db1[v];
    }
#pragma unroll
    for (int v = 0; v < 8; ++v) {
      float mv = rmax16(fmaxf(e0[v], e1[v]));
      float mn = fmaxf(m_old[v], mv);
      alpha[v] = __expf(m_old[v] - mn);
      m_old[v] = mn;
      e0[v] = __expf(e0[v] - mn);
      e1[v] = __expf(e1[v] - mn);
      float rs = rsum16(e0[v] + e1[v]);
      l_sum[v] = l_sum[v] * alpha[v] + rs;
    }

    // --- P*d -> LDS (D-layout -> A-layout transpose within the wave) ---
#pragma unroll
    for (int v = 0; v < 8; ++v) {
      int nloc = (lane >> 4) * 8 + v;
      pl[nloc * PROW + col]      = (_Float16)(e0[v] * db0[v]);
      pl[nloc * PROW + 16 + col] = (_Float16)(e1[v] * db1[v]);
    }
    asm volatile("s_wait_dscnt 0" ::: "memory");
    v16h ap = load_frag_h(pl + col * PROW, base0);

    // --- rescale accumulators, then P @ V over four d-tiles (from LDS) ---
#pragma unroll
    for (int j = 0; j < 4; ++j)
#pragma unroll
      for (int v = 0; v < 8; ++v) o[j][v] *= alpha[v];
#pragma unroll
    for (int j = 0; j < 4; ++j) {
      v16h bv = frag16(Vb + cur * VBUFSZ + (j * 16 + col) * VROW + kb);
      o[j] = wmma_f16(ap, bv, o[j]);
    }

    __syncthreads();     // all waves done with 'cur' before it is overwritten
  }

  // --- finalize: divide by Z, store f16 merged-head layout ---
#pragma unroll
  for (int v = 0; v < 8; ++v) l_sum[v] = 1.0f / l_sum[v];
#pragma unroll
  for (int j = 0; j < 4; ++j) {
    int ec = h * DHH + j * 16 + col;
#pragma unroll
    for (int v = 0; v < 8; ++v) {
      int r = n0 + (lane >> 4) * 8 + v;
      Xb[(size_t)(b * NNQ + r) * EE + ec] = (_Float16)(o[j][v] * l_sum[v]);
    }
  }
}

// ---------------- output projection: out = X * Wp^T + bp (fp32 out) ---------
__global__ void oproj_kernel(const _Float16* __restrict__ Xb,
                             const _Float16* __restrict__ W,
                             const float* __restrict__ bias,
                             float* __restrict__ out) {
  const int lane  = threadIdx.x & 31;
  const int wave  = threadIdx.x >> 5;
  const int mtile = blockIdx.x * 8 + wave;
  const int row0  = mtile * 16;
  const int fs    = blockIdx.y * 64;
  const int base0 = (lane >> 4) * 8;
  const int kb    = (lane >> 4) * 16;
  const int colA  = lane & 15;

  v8f z = {};
  v8f c[4];
  c[0] = z; c[1] = z; c[2] = z; c[3] = z;

  const _Float16* arow = Xb + (size_t)(row0 + colA) * EE;

  for (int k = 0; k < EE; k += 32) {
    v16h a = load_frag_h(arow + k, base0);
#pragma unroll
    for (int ft = 0; ft < 4; ++ft) {
      int f = fs + ft * 16 + colA;
      v16h b = *(const v16h*)(W + (size_t)f * EE + k + kb);
      c[ft] = wmma_f16(a, b, c[ft]);
    }
  }

#pragma unroll
  for (int ft = 0; ft < 4; ++ft) {
    int f = fs + ft * 16 + colA;
    float bv = bias[f];
#pragma unroll
    for (int v = 0; v < 8; ++v) {
      int r = row0 + (lane >> 4) * 8 + v;
      out[(size_t)r * EE + f] = c[ft][v] + bv;
    }
  }
}

extern "C" void kernel_launch(void* const* d_in, const int* in_sizes, int n_in,
                              void* d_out, int out_size, void* d_ws, size_t ws_size,
                              hipStream_t stream) {
  (void)in_sizes; (void)n_in; (void)out_size; (void)ws_size;
  const float* q  = (const float*)d_in[0];
  const float* kk = (const float*)d_in[1];
  const float* v  = (const float*)d_in[2];
  const float* dm = (const float*)d_in[3];
  const float* Wq = (const float*)d_in[4];
  const float* bq = (const float*)d_in[5];
  const float* Wk = (const float*)d_in[6];
  const float* bk = (const float*)d_in[7];
  const float* Wv = (const float*)d_in[8];
  const float* bv = (const float*)d_in[9];
  const float* Wp = (const float*)d_in[10];
  const float* bp = (const float*)d_in[11];
  float* out = (float*)d_out;

  // workspace carve-up (f16 elements). total = 2 MB weights + 64 MB tensors.
  _Float16* Wq16 = (_Float16*)d_ws;
  _Float16* Wk16 = Wq16 + 262144;
  _Float16* Wv16 = Wk16 + 262144;
  _Float16* Wp16 = Wv16 + 262144;
  _Float16* Q16  = Wp16 + 262144;         // [b*N][E]  plain
  _Float16* K16  = Q16 + 8388608;         // [b*N][E]  plain
  _Float16* Vt16 = K16 + 8388608;         // [(b*H+h)*DH + d][N]  transposed
  _Float16* Xb16 = Vt16 + 8388608;        // [b*N][E]  attention output

  wcvt_kernel<<<4096, 256, 0, stream>>>(Wq, Wk, Wv, Wp, Wq16);
  proj_kernel<<<dim3(128, 8), 256, 0, stream>>>(q,  Wq16, bq, Q16, 0);
  proj_kernel<<<dim3(128, 8), 256, 0, stream>>>(kk, Wk16, bk, K16, 0);
  proj_kernel<<<dim3(128, 8), 256, 0, stream>>>(v,  Wv16, bv, Vt16, 1);
  attn_kernel<<<dim3(128, 8), 256, 0, stream>>>(Q16, K16, Vt16, dm, Xb16);
  oproj_kernel<<<dim3(128, 8), 256, 0, stream>>>(Xb16, Wp16, bp, out);
}
